// CrossAttentionSequencePool_37830071943283
// MI455X (gfx1250) — compile-verified, matrix operand order checked by reading, not run on hardware
//
#include <hip/hip_runtime.h>
#include <hip/hip_bf16.h>
#include <stdint.h>

// ---------------- problem constants ----------------
#define INPUT_DIM  256
#define HIDDEN_DIM 256
#define M_Q        1024
#define N_K        32768
#define X_DIM      (3 * INPUT_DIM)     // 768
#define N_SEG      256
#define SEG_LEN    (N_K / N_SEG)       // 128

typedef __attribute__((ext_vector_type(16))) __bf16 v16bf;
typedef __attribute__((ext_vector_type(8)))  __bf16 v8bf;
typedef __attribute__((ext_vector_type(8)))  float  v8f;

union FragB16 { v16bf v; v8bf h[2]; };

__device__ __forceinline__ uint16_t f32_to_bf16_rne(float f) {
    union { float f; uint32_t u; } v; v.f = f;
    uint32_t u = v.u;
    uint32_t r = (u + 0x7FFFu + ((u >> 16) & 1u)) >> 16;
    return (uint16_t)r;
}

#define NEG_HUGE (-3.402823466e38f)

// ---------------- elementwise f32 -> bf16 convert ----------------
__global__ void cvt_f32_bf16(const float* __restrict__ in,
                             uint16_t* __restrict__ out, int n) {
    int i = blockIdx.x * blockDim.x + threadIdx.x;
    int stride = gridDim.x * blockDim.x;
    for (; i < n; i += stride) out[i] = f32_to_bf16_rne(in[i]);
}

// ---------------- scan phase 1: per-segment column max ----------------
// grid = N_SEG blocks, block = 256 threads (one per column)
__global__ void seg_max_kernel(const float* __restrict__ key,
                               float* __restrict__ segmax) {
    int col = threadIdx.x;
    int seg = blockIdx.x;
    float m = NEG_HUGE;
    const float* p = key + (size_t)seg * SEG_LEN * INPUT_DIM + col;
    #pragma unroll 4
    for (int r = 0; r < SEG_LEN; ++r)
        m = fmaxf(m, p[(size_t)r * INPUT_DIM]);
    segmax[seg * INPUT_DIM + col] = m;
}

// ---------------- scan phase 2: exclusive prefix/suffix over segments ----
// single block of 256 threads (one per column)
__global__ void seg_scan_kernel(const float* __restrict__ segmax,
                                float* __restrict__ segpre,
                                float* __restrict__ segsuf) {
    int col = threadIdx.x;
    float run = NEG_HUGE;
    for (int s = 0; s < N_SEG; ++s) {
        segpre[s * INPUT_DIM + col] = run;
        run = fmaxf(run, segmax[s * INPUT_DIM + col]);
    }
    run = NEG_HUGE;
    for (int s = N_SEG - 1; s >= 0; --s) {
        segsuf[s * INPUT_DIM + col] = run;
        run = fmaxf(run, segmax[s * INPUT_DIM + col]);
    }
}

// ---------------- scan phase 3: build bf16 X = [key | before | after] ----
// grid = N_SEG blocks, block = 256 threads (one per column)
__global__ void build_x_kernel(const float* __restrict__ key,
                               const float* __restrict__ segpre,
                               const float* __restrict__ segsuf,
                               uint16_t* __restrict__ xbf) {
    int col = threadIdx.x;
    int seg = blockIdx.x;
    // forward: key + exclusive prefix max (before)
    float run = segpre[seg * INPUT_DIM + col];
    for (int r = 0; r < SEG_LEN; ++r) {
        int row = seg * SEG_LEN + r;
        float kv = key[(size_t)row * INPUT_DIM + col];
        float before = (row == 0) ? 0.0f : run;
        uint16_t* xr = xbf + (size_t)row * X_DIM;
        xr[col]             = f32_to_bf16_rne(kv);
        xr[INPUT_DIM + col] = f32_to_bf16_rne(before);
        run = fmaxf(run, kv);
    }
    // backward: exclusive suffix max (after)
    run = segsuf[seg * INPUT_DIM + col];
    for (int r = SEG_LEN - 1; r >= 0; --r) {
        int row = seg * SEG_LEN + r;
        float kv = key[(size_t)row * INPUT_DIM + col];
        float after = (row == N_K - 1) ? 0.0f : run;
        xbf[(size_t)row * X_DIM + 2 * INPUT_DIM + col] = f32_to_bf16_rne(after);
        run = fmaxf(run, kv);
    }
}

// ---------------- generic bf16 WMMA GEMM ----------------
// C[M,N] = act( scale * (A[M,K] @ Wt[N,K]^T) + bias[N] )
// A row-major bf16, Wt row-major bf16 ([out,in] weight layout).
// Each wave computes a 32x64 output tile: 2 A-frags x 4 B-frags ->
// 8 WMMAs per K-step. All six fragments are loaded into distinct
// registers BEFORE the WMMA burst so the scheduler can pipeline the
// next iteration's loads under the current WMMAs.
template <bool RELU, bool OUT_BF16>
__global__ void gemm_bf16_wmma(const uint16_t* __restrict__ A,
                               const uint16_t* __restrict__ Wt,
                               const float* __restrict__ bias,
                               void* __restrict__ Cout,
                               int M, int N, int K, float scale) {
    const int lane = threadIdx.x & 31;
    const int wib  = threadIdx.x >> 5;                 // wave in block (0..7)
    const int wid  = blockIdx.x * (blockDim.x >> 5) + wib;

    const int nTilesM = M >> 5;                        // 32-row tiles
    const int nTilesN = N >> 6;                        // 64-col tiles
    if (wid >= nTilesM * nTilesN) return;              // wave-uniform exit

    const int tm = wid % nTilesM;
    const int tn = wid / nTilesM;
    const int m0 = tm << 5;
    const int n0 = tn << 6;

    const int lm   = lane & 15;
    const int half = lane >> 4;
    const int koff = half * 8;                         // lane K sub-offset

    v8f acc[2][4];
    #pragma unroll
    for (int i = 0; i < 2; ++i)
        #pragma unroll
        for (int j = 0; j < 4; ++j) acc[i][j] = (v8f){};

    const uint16_t* pa0 = A + (size_t)(m0 + 0  + lm) * K + koff;
    const uint16_t* pa1 = A + (size_t)(m0 + 16 + lm) * K + koff;
    const uint16_t* pb0 = Wt + (size_t)(n0 + 0 * 16 + lm) * K + koff;
    const uint16_t* pb1 = Wt + (size_t)(n0 + 1 * 16 + lm) * K + koff;
    const uint16_t* pb2 = Wt + (size_t)(n0 + 2 * 16 + lm) * K + koff;
    const uint16_t* pb3 = Wt + (size_t)(n0 + 3 * 16 + lm) * K + koff;

    for (int k0 = 0; k0 < K; k0 += 32) {
        FragB16 a0, a1, b0, b1, b2, b3;
        // ---- load phase: all fragments into distinct registers ----
        a0.h[0] = *(const v8bf*)(pa0 + k0);
        a0.h[1] = *(const v8bf*)(pa0 + k0 + 16);
        a1.h[0] = *(const v8bf*)(pa1 + k0);
        a1.h[1] = *(const v8bf*)(pa1 + k0 + 16);
        b0.h[0] = *(const v8bf*)(pb0 + k0);
        b0.h[1] = *(const v8bf*)(pb0 + k0 + 16);
        b1.h[0] = *(const v8bf*)(pb1 + k0);
        b1.h[1] = *(const v8bf*)(pb1 + k0 + 16);
        b2.h[0] = *(const v8bf*)(pb2 + k0);
        b2.h[1] = *(const v8bf*)(pb2 + k0 + 16);
        b3.h[0] = *(const v8bf*)(pb3 + k0);
        b3.h[1] = *(const v8bf*)(pb3 + k0 + 16);
        __builtin_prefetch(pa0 + k0 + 32, 0, 1);       // global_prefetch next A

        // ---- compute phase: 8 WMMAs ----
        acc[0][0] = __builtin_amdgcn_wmma_f32_16x16x32_bf16(false, a0.v, false, b0.v,
                                                            (short)0, acc[0][0], false, false);
        acc[1][0] = __builtin_amdgcn_wmma_f32_16x16x32_bf16(false, a1.v, false, b0.v,
                                                            (short)0, acc[1][0], false, false);
        acc[0][1] = __builtin_amdgcn_wmma_f32_16x16x32_bf16(false, a0.v, false, b1.v,
                                                            (short)0, acc[0][1], false, false);
        acc[1][1] = __builtin_amdgcn_wmma_f32_16x16x32_bf16(false, a1.v, false, b1.v,
                                                            (short)0, acc[1][1], false, false);
        acc[0][2] = __builtin_amdgcn_wmma_f32_16x16x32_bf16(false, a0.v, false, b2.v,
                                                            (short)0, acc[0][2], false, false);
        acc[1][2] = __builtin_amdgcn_wmma_f32_16x16x32_bf16(false, a1.v, false, b2.v,
                                                            (short)0, acc[1][2], false, false);
        acc[0][3] = __builtin_amdgcn_wmma_f32_16x16x32_bf16(false, a0.v, false, b3.v,
                                                            (short)0, acc[0][3], false, false);
        acc[1][3] = __builtin_amdgcn_wmma_f32_16x16x32_bf16(false, a1.v, false, b3.v,
                                                            (short)0, acc[1][3], false, false);
    }

    // Write out: acc[i][j] VGPR r holds row m0 + 16*i + 8*half + r,
    // col n0 + 16*j + lm.
    #pragma unroll
    for (int i = 0; i < 2; ++i) {
        #pragma unroll
        for (int j = 0; j < 4; ++j) {
            const int n = n0 + 16 * j + lm;
            const float bv = bias ? bias[n] : 0.0f;
            #pragma unroll
            for (int r = 0; r < 8; ++r) {
                const int m = m0 + 16 * i + half * 8 + r;
                float val = acc[i][j][r] * scale + bv;
                if (RELU) val = fmaxf(val, 0.0f);
                if (OUT_BF16)
                    ((uint16_t*)Cout)[(size_t)m * N + n] = f32_to_bf16_rne(val);
                else
                    ((float*)Cout)[(size_t)m * N + n] = val;
            }
        }
    }
}

// ---------------- launcher ----------------
extern "C" void kernel_launch(void* const* d_in, const int* in_sizes, int n_in,
                              void* d_out, int out_size, void* d_ws, size_t ws_size,
                              hipStream_t stream) {
    const float* query = (const float*)d_in[0];   // [1024,256]
    const float* key   = (const float*)d_in[1];   // [32768,256]
    const float* q1_w  = (const float*)d_in[2];   // [256,256]
    const float* q1_b  = (const float*)d_in[3];   // [256]
    const float* q2_w  = (const float*)d_in[4];   // [256,256]
    const float* q2_b  = (const float*)d_in[5];   // [256]
    const float* k1_w  = (const float*)d_in[6];   // [256,768]
    const float* k1_b  = (const float*)d_in[7];   // [256]
    const float* k2_w  = (const float*)d_in[8];   // [256,256]
    const float* k2_b  = (const float*)d_in[9];   // [256]
    float* out = (float*)d_out;                   // [1024,32768]

    // workspace carve-up (256B aligned)
    char* base = (char*)d_ws;
    size_t off = 0;
    auto carve = [&](size_t bytes) -> char* {
        char* p = base + off;
        off = (off + bytes + 255) & ~(size_t)255;
        return p;
    };
    uint16_t* xbf   = (uint16_t*)carve((size_t)N_K * X_DIM * 2);        // 50.3 MB
    uint16_t* k1w   = (uint16_t*)carve((size_t)HIDDEN_DIM * X_DIM * 2);
    uint16_t* k2w   = (uint16_t*)carve((size_t)HIDDEN_DIM * HIDDEN_DIM * 2);
    uint16_t* q1w   = (uint16_t*)carve((size_t)HIDDEN_DIM * INPUT_DIM * 2);
    uint16_t* q2w   = (uint16_t*)carve((size_t)HIDDEN_DIM * HIDDEN_DIM * 2);
    uint16_t* qbf   = (uint16_t*)carve((size_t)M_Q * INPUT_DIM * 2);
    uint16_t* h1k   = (uint16_t*)carve((size_t)N_K * HIDDEN_DIM * 2);   // 16.8 MB
    uint16_t* xkey  = (uint16_t*)carve((size_t)N_K * HIDDEN_DIM * 2);   // 16.8 MB
    uint16_t* hq    = (uint16_t*)carve((size_t)M_Q * HIDDEN_DIM * 2);
    uint16_t* xq    = (uint16_t*)carve((size_t)M_Q * HIDDEN_DIM * 2);
    float*    smax  = (float*)carve((size_t)N_SEG * INPUT_DIM * 4);
    float*    spre  = (float*)carve((size_t)N_SEG * INPUT_DIM * 4);
    float*    ssuf  = (float*)carve((size_t)N_SEG * INPUT_DIM * 4);

    // 1) convert weights + query to bf16
    cvt_f32_bf16<<<192, 256, 0, stream>>>(k1_w, k1w, HIDDEN_DIM * X_DIM);
    cvt_f32_bf16<<<64,  256, 0, stream>>>(k2_w, k2w, HIDDEN_DIM * HIDDEN_DIM);
    cvt_f32_bf16<<<64,  256, 0, stream>>>(q1_w, q1w, HIDDEN_DIM * INPUT_DIM);
    cvt_f32_bf16<<<64,  256, 0, stream>>>(q2_w, q2w, HIDDEN_DIM * HIDDEN_DIM);
    cvt_f32_bf16<<<256, 256, 0, stream>>>(query, qbf, M_Q * INPUT_DIM);

    // 2) hierarchical max-scan + build X bf16
    seg_max_kernel <<<N_SEG, 256, 0, stream>>>(key, smax);
    seg_scan_kernel<<<1,     256, 0, stream>>>(smax, spre, ssuf);
    build_x_kernel <<<N_SEG, 256, 0, stream>>>(key, spre, ssuf, xbf);

    // 3) key MLP
    {   // layer1: relu(X @ k1_w^T + k1_b) -> h1k   M=32768 N=256 K=768
        int waves = (N_K / 32) * (HIDDEN_DIM / 64);
        gemm_bf16_wmma<true, true><<<(waves + 7) / 8, 256, 0, stream>>>(
            xbf, k1w, k1_b, h1k, N_K, HIDDEN_DIM, X_DIM, 1.0f);
    }
    {   // layer2: h1k @ k2_w^T + k2_b -> xkey      M=32768 N=256 K=256
        int waves = (N_K / 32) * (HIDDEN_DIM / 64);
        gemm_bf16_wmma<false, true><<<(waves + 7) / 8, 256, 0, stream>>>(
            h1k, k2w, k2_b, xkey, N_K, HIDDEN_DIM, HIDDEN_DIM, 1.0f);
    }

    // 4) query MLP
    {   // layer1: relu(Q @ q1_w^T + q1_b) -> hq    M=1024 N=256 K=256
        int waves = (M_Q / 32) * (HIDDEN_DIM / 64);
        gemm_bf16_wmma<true, true><<<(waves + 7) / 8, 256, 0, stream>>>(
            qbf, q1w, q1_b, hq, M_Q, HIDDEN_DIM, INPUT_DIM, 1.0f);
    }
    {   // layer2: hq @ q2_w^T + q2_b -> xq         M=1024 N=256 K=256
        int waves = (M_Q / 32) * (HIDDEN_DIM / 64);
        gemm_bf16_wmma<false, true><<<(waves + 7) / 8, 256, 0, stream>>>(
            hq, q2w, q2_b, xq, M_Q, HIDDEN_DIM, HIDDEN_DIM, 1.0f);
    }

    // 5) final: (xq @ xkey^T) / 16 -> out (f32)    M=1024 N=32768 K=256
    {
        int waves = (M_Q / 32) * (N_K / 64);
        gemm_bf16_wmma<false, false><<<(waves + 7) / 8, 256, 0, stream>>>(
            xq, xkey, nullptr, out, M_Q, N_K, HIDDEN_DIM, 0.0625f);
    }
}
